// CTransformer_26010321945015
// MI455X (gfx1250) — matrix operand order repaired and checked by reading, not compiled, not run on hardware
//
#include <hip/hip_runtime.h>
#include <math.h>

// ---------------- problem constants ----------------
#define BATCH   64
#define CH      256
#define HH      32
#define WW      32
#define HWSZ    (HH*WW)            // 1024
#define TOKENS  (BATCH*HWSZ)       // 65536
#define HEADS   4
#define HD      64
#define HIDDEN  1024
#define DEPTH   2
#define WIN     64                 // HSP*WSP tokens per window (consecutive in token-major)
#define NWIN    (TOKENS/WIN)       // 1024
#define ATTN_SCALE 0.125f          // HD^-0.5
#define LNEPS   1e-5f

// ---------------- vector types ----------------
typedef __attribute__((ext_vector_type(16))) __bf16 v16bf;
typedef __attribute__((ext_vector_type(8)))  __bf16 v8bf;
typedef __attribute__((ext_vector_type(4)))  __bf16 v4bf;
typedef __attribute__((ext_vector_type(8)))  float  v8f;

__device__ inline v8f zero8() {
    v8f z;
#pragma unroll
    for (int i = 0; i < 8; ++i) z[i] = 0.0f;
    return z;
}

// A fragment (16x32 bf16, M x K). Contiguous layout: lane<16 holds K 0-15,
// lane>=16 holds K 16-31; m = m0 + (lane&15). Two 16B LDS loads.
__device__ inline v16bf lds_a_frag(const __bf16* lds, int m0, int k0, int stride) {
    const int lane = threadIdx.x & 31;
    const int m    = m0 + (lane & 15);
    const int half = lane >> 4;
    const v8bf a0 = *(const v8bf*)(lds + m * stride + k0 + half * 16);
    const v8bf a1 = *(const v8bf*)(lds + m * stride + k0 + half * 16 + 8);
    v16bf r;
#pragma unroll
    for (int i = 0; i < 8; ++i) { r[i] = a0[i]; r[8 + i] = a1[i]; }
    return r;
}

// B fragment (32x16 bf16, K x N), LDS staged n-major: lane column n = n0+(lane&15),
// lane half selects K half -> 16 contiguous K values = one 32B LDS load.
__device__ inline v16bf lds_b_frag(const __bf16* lds, int n0, int k0, int stride) {
    const int lane = threadIdx.x & 31;
    const int n    = n0 + (lane & 15);
    const int half = lane >> 4;
    return *(const v16bf*)(lds + n * stride + k0 + half * 16);
}

__device__ inline float gelu_exact(float x) {
    return 0.5f * x * (1.0f + erff(x * 0.70710678118654752f));
}

// ---------------------------------------------------------------------------
// Generic token-major GEMM: out[M,N] = (A[M,K](+Aadd bcast over batch)) * W^T
// + bias, optional GELU / residual (compile-time specialized). W stored (N,K)
// row-major bf16 (== B operand n-major). Block: 256 thr = 8 waves (2x4),
// tile 128x256, K-step 64 (LDS: 16KB A + 32KB B).
// ---------------------------------------------------------------------------
template<bool HAS_ADD, bool DO_GELU, bool HAS_RES>
__global__ __launch_bounds__(256)
void gemm_bf16_wmma(const float* __restrict__ A,
                    const float* __restrict__ Aadd,
                    const __bf16* __restrict__ Wb,
                    const float* __restrict__ bias,
                    const float* __restrict__ res,
                    float* __restrict__ out,
                    int M, int N, int K)
{
    __shared__ __bf16 Alds[128 * 64];   // m-major
    __shared__ __bf16 Blds[256 * 64];   // n-major

    const int tid  = threadIdx.x;
    const int lane = tid & 31;
    const int wave = tid >> 5;
    const int wm   = wave >> 2;         // 0..1
    const int wn   = wave & 3;          // 0..3
    const long baseM = (long)blockIdx.x * 128;
    const long baseN = (long)blockIdx.y * 256;

    v8f acc[16];
#pragma unroll
    for (int i = 0; i < 16; ++i) acc[i] = zero8();

    for (int k0 = 0; k0 < K; k0 += 64) {
        // ---- stage A (128x64 f32 -> bf16); 8 float4 per thread ----
#pragma unroll
        for (int i = 0; i < 8; ++i) {
            const int idx = i * 256 + tid;
            const int row = idx >> 4;           // 16 float4 per 64-wide row
            const int c4  = (idx & 15) * 4;
            const long gr = baseM + row;
            float4 v = *(const float4*)(A + gr * (long)K + k0 + c4);
            if (HAS_ADD) {
                // positional embedding broadcast over batch: period HWSZ (pow2)
                const float4 e = *(const float4*)(Aadd + (gr & (HWSZ - 1)) * (long)K + k0 + c4);
                v.x += e.x; v.y += e.y; v.z += e.z; v.w += e.w;
            }
            v4bf bv;
            bv[0] = (__bf16)v.x; bv[1] = (__bf16)v.y;
            bv[2] = (__bf16)v.z; bv[3] = (__bf16)v.w;
            *(v4bf*)(Alds + row * 64 + c4) = bv;
        }
        // ---- stage B (256x64 bf16, one 128B row per thread) ----
        {
            const __bf16* src = Wb + (baseN + tid) * (long)K + k0;
#pragma unroll
            for (int j = 0; j < 8; ++j)
                ((uint4*)(Blds + tid * 64))[j] = ((const uint4*)src)[j];
        }
        // prefetch next A k-slice (global_prefetch_b8)
        if (k0 + 64 < K)
            __builtin_prefetch(A + (baseM + (tid & 127)) * (long)K + k0 + 64, 0, 1);
        __syncthreads();

#pragma unroll
        for (int kk = 0; kk < 64; kk += 32) {
            v16bf af[4], bfg[4];
#pragma unroll
            for (int mi = 0; mi < 4; ++mi) af[mi]  = lds_a_frag(Alds, wm * 64 + mi * 16, kk, 64);
#pragma unroll
            for (int ni = 0; ni < 4; ++ni) bfg[ni] = lds_b_frag(Blds, wn * 64 + ni * 16, kk, 64);
#pragma unroll
            for (int mi = 0; mi < 4; ++mi)
#pragma unroll
                for (int ni = 0; ni < 4; ++ni)
                    acc[mi * 4 + ni] = __builtin_amdgcn_wmma_f32_16x16x32_bf16(
                        false, af[mi], false, bfg[ni], (short)0, acc[mi * 4 + ni], false, false);
        }
        __syncthreads();
    }

    // ---- epilogue: C/D layout lane -> (col = lane&15, rows = (lane>>4)*8 + r) ----
    const int colOff = lane & 15;
    const int rowOff = (lane >> 4) * 8;
#pragma unroll
    for (int mi = 0; mi < 4; ++mi) {
#pragma unroll
        for (int ni = 0; ni < 4; ++ni) {
            const long n = baseN + wn * 64 + ni * 16 + colOff;
            const float bn = bias[n];
#pragma unroll
            for (int r = 0; r < 8; ++r) {
                const long m = baseM + wm * 64 + mi * 16 + rowOff + r;
                float v = acc[mi * 4 + ni][r] + bn;
                if (DO_GELU) v = gelu_exact(v);
                if (HAS_RES) v += res[m * N + n];
                out[m * N + n] = v;
            }
        }
    }
}

// ---------------------------------------------------------------------------
// Windowed attention: one block per (window, head); 64 threads = 2 waves.
// Window = 64 consecutive tokens; head = 64-channel slice. S = scale*Q*K^T,
// softmax, O = P*V. All matmuls via v_wmma_f32_16x16x32_bf16.
// ---------------------------------------------------------------------------
__global__ __launch_bounds__(64)
void attn_wmma(const float* __restrict__ q,
               const float* __restrict__ k,
               const float* __restrict__ v,
               float* __restrict__ o)
{
    __shared__ __bf16 Qs[64 * 64];   // Q (scaled), later reused for P (A-layout, m-major)
    __shared__ __bf16 Ks[64 * 64];   // K token-major == B n-major for Q*K^T
    __shared__ __bf16 Vt[64 * 64];   // V^T: [channel][token] == B n-major for P*V
    __shared__ float  Ss[64 * 64];   // scores

    const int tid  = threadIdx.x;
    const int lane = tid & 31;
    const int wv   = tid >> 5;                 // 0..1
    const long tb  = (long)blockIdx.x * WIN;   // token base of window
    const int  cb  = blockIdx.y * HD;          // channel base of head

    // ---- load Q,K,V head slices (one 64-ch row per thread) ----
    {
        const int m = tid;
        const float* qr = q + (tb + m) * CH + cb;
        const float* kr = k + (tb + m) * CH + cb;
        const float* vr = v + (tb + m) * CH + cb;
#pragma unroll
        for (int c4 = 0; c4 < 16; ++c4) {
            const float4 qv = *(const float4*)(qr + c4 * 4);
            const float4 kv = *(const float4*)(kr + c4 * 4);
            const float4 vv = *(const float4*)(vr + c4 * 4);
            v4bf qb, kb;
            qb[0] = (__bf16)(qv.x * ATTN_SCALE); qb[1] = (__bf16)(qv.y * ATTN_SCALE);
            qb[2] = (__bf16)(qv.z * ATTN_SCALE); qb[3] = (__bf16)(qv.w * ATTN_SCALE);
            kb[0] = (__bf16)kv.x; kb[1] = (__bf16)kv.y;
            kb[2] = (__bf16)kv.z; kb[3] = (__bf16)kv.w;
            *(v4bf*)(Qs + m * 64 + c4 * 4) = qb;
            *(v4bf*)(Ks + m * 64 + c4 * 4) = kb;
            Vt[(c4 * 4 + 0) * 64 + m] = (__bf16)vv.x;
            Vt[(c4 * 4 + 1) * 64 + m] = (__bf16)vv.y;
            Vt[(c4 * 4 + 2) * 64 + m] = (__bf16)vv.z;
            Vt[(c4 * 4 + 3) * 64 + m] = (__bf16)vv.w;
        }
    }
    __syncthreads();

    const int colOff = lane & 15;
    const int rowOff = (lane >> 4) * 8;
    const int m0w    = wv * 32;      // wave's 32-row slab

    // ---- S = (scale*Q) @ K^T ----
    {
        v8f acc[8];
#pragma unroll
        for (int i = 0; i < 8; ++i) acc[i] = zero8();
#pragma unroll
        for (int ki = 0; ki < 2; ++ki) {
            v16bf af[2], bfg[4];
#pragma unroll
            for (int mi = 0; mi < 2; ++mi) af[mi]  = lds_a_frag(Qs, m0w + mi * 16, ki * 32, 64);
#pragma unroll
            for (int ni = 0; ni < 4; ++ni) bfg[ni] = lds_b_frag(Ks, ni * 16, ki * 32, 64);
#pragma unroll
            for (int mi = 0; mi < 2; ++mi)
#pragma unroll
                for (int ni = 0; ni < 4; ++ni)
                    acc[mi * 4 + ni] = __builtin_amdgcn_wmma_f32_16x16x32_bf16(
                        false, af[mi], false, bfg[ni], (short)0, acc[mi * 4 + ni], false, false);
        }
#pragma unroll
        for (int mi = 0; mi < 2; ++mi)
#pragma unroll
            for (int ni = 0; ni < 4; ++ni)
#pragma unroll
                for (int r = 0; r < 8; ++r)
                    Ss[(m0w + mi * 16 + rowOff + r) * 64 + ni * 16 + colOff] = acc[mi * 4 + ni][r];
    }
    __syncthreads();

    // ---- row softmax; P (bf16) overwrites Qs in A-layout (m-major) ----
    {
        const int m = tid;
        float* row = Ss + m * 64;
        float mx = -3.4e38f;
#pragma unroll
        for (int j = 0; j < 64; ++j) mx = fmaxf(mx, row[j]);
        float s = 0.0f;
#pragma unroll
        for (int j = 0; j < 64; ++j) { const float e = __expf(row[j] - mx); row[j] = e; s += e; }
        const float inv = 1.0f / s;
#pragma unroll
        for (int j4 = 0; j4 < 16; ++j4) {
            v4bf p;
            p[0] = (__bf16)(row[j4 * 4 + 0] * inv);
            p[1] = (__bf16)(row[j4 * 4 + 1] * inv);
            p[2] = (__bf16)(row[j4 * 4 + 2] * inv);
            p[3] = (__bf16)(row[j4 * 4 + 3] * inv);
            *(v4bf*)(Qs + m * 64 + j4 * 4) = p;
        }
    }
    __syncthreads();

    // ---- O = P @ V ----
    {
        v8f acc[8];
#pragma unroll
        for (int i = 0; i < 8; ++i) acc[i] = zero8();
#pragma unroll
        for (int ki = 0; ki < 2; ++ki) {
            v16bf af[2], bfg[4];
#pragma unroll
            for (int mi = 0; mi < 2; ++mi) af[mi]  = lds_a_frag(Qs, m0w + mi * 16, ki * 32, 64);
#pragma unroll
            for (int ni = 0; ni < 4; ++ni) bfg[ni] = lds_b_frag(Vt, ni * 16, ki * 32, 64);
#pragma unroll
            for (int mi = 0; mi < 2; ++mi)
#pragma unroll
                for (int ni = 0; ni < 4; ++ni)
                    acc[mi * 4 + ni] = __builtin_amdgcn_wmma_f32_16x16x32_bf16(
                        false, af[mi], false, bfg[ni], (short)0, acc[mi * 4 + ni], false, false);
        }
#pragma unroll
        for (int mi = 0; mi < 2; ++mi)
#pragma unroll
            for (int ni = 0; ni < 4; ++ni)
#pragma unroll
                for (int r = 0; r < 8; ++r) {
                    const long m = tb + m0w + mi * 16 + rowOff + r;
                    o[m * CH + cb + ni * 16 + colOff] = acc[mi * 4 + ni][r];
                }
    }
}

// ---------------------------------------------------------------------------
// LePE: depthwise 3x3 conv over each 2x32 window (zero-padded), + bias,
// accumulated into attention output. o[t,ch] += conv + blp[ch].
// ---------------------------------------------------------------------------
__global__ __launch_bounds__(256)
void lepe_add(const float* __restrict__ v, const float* __restrict__ w,
              const float* __restrict__ b, float* __restrict__ o)
{
    const long idx = (long)blockIdx.x * 256 + threadIdx.x;   // over TOKENS*CH
    const int  ch  = (int)(idx & (CH - 1));
    const long t   = idx >> 8;
    const long g   = t >> 6;          // window
    const int  l   = (int)(t & 63);
    const int  r   = l >> 5;          // 0..1
    const int  c   = l & 31;          // 0..31
    float s = 0.0f;
#pragma unroll
    for (int dr = 0; dr < 3; ++dr) {
        const int rr = r + dr - 1;
        if (rr < 0 || rr > 1) continue;
#pragma unroll
        for (int dc = 0; dc < 3; ++dc) {
            const int cc = c + dc - 1;
            if (cc < 0 || cc > 31) continue;
            s += w[ch * 9 + dr * 3 + dc] * v[((g << 6) + rr * 32 + cc) * CH + ch];
        }
    }
    o[idx] += s + b[ch];
}

// ---------------------------------------------------------------------------
// y = x + o ; LayerNorm(y)*gamma+beta -> yn.  One wave32 per token.
// ---------------------------------------------------------------------------
__global__ __launch_bounds__(256)
void resid_layernorm(const float* __restrict__ x, const float* __restrict__ o,
                     const float* __restrict__ g, const float* __restrict__ be,
                     float* __restrict__ y, float* __restrict__ yn)
{
    const int tid  = threadIdx.x;
    const int lane = tid & 31;
    const int wv   = tid >> 5;
    const long t   = (long)blockIdx.x * 8 + wv;
    const long base = t * CH + lane * 8;

    float yv[8];
    float s = 0.0f, sq = 0.0f;
#pragma unroll
    for (int i = 0; i < 2; ++i) {
        const float4 xa = *(const float4*)(x + base + i * 4);
        const float4 oa = *(const float4*)(o + base + i * 4);
        const float4 ya = make_float4(xa.x + oa.x, xa.y + oa.y, xa.z + oa.z, xa.w + oa.w);
        yv[i * 4 + 0] = ya.x; yv[i * 4 + 1] = ya.y; yv[i * 4 + 2] = ya.z; yv[i * 4 + 3] = ya.w;
        s  += ya.x + ya.y + ya.z + ya.w;
        sq += ya.x * ya.x + ya.y * ya.y + ya.z * ya.z + ya.w * ya.w;
        *(float4*)(y + base + i * 4) = ya;
    }
#pragma unroll
    for (int off = 16; off > 0; off >>= 1) {
        s  += __shfl_xor(s,  off, 32);
        sq += __shfl_xor(sq, off, 32);
    }
    const float mu   = s * (1.0f / CH);
    const float var  = sq * (1.0f / CH) - mu * mu;
    const float rstd = rsqrtf(var + LNEPS);
#pragma unroll
    for (int i = 0; i < 8; ++i) {
        const int ch = lane * 8 + i;
        yn[base + i] = (yv[i] - mu) * rstd * g[ch] + be[ch];
    }
}

// ---------------------------------------------------------------------------
// Layout helpers
// ---------------------------------------------------------------------------
__global__ void chw_to_tc(const float* __restrict__ in, float* __restrict__ out, long total)
{   // (S,C,HW) -> (S,HW,C)
    const long idx = (long)blockIdx.x * 256 + threadIdx.x;
    if (idx >= total) return;
    const long s  = idx / ((long)HWSZ * CH);
    const long r  = idx - s * ((long)HWSZ * CH);
    const long hw = r / CH;
    const long c  = r - hw * CH;
    out[idx] = in[(s * CH + c) * HWSZ + hw];
}

__global__ void tc_to_chw(const float* __restrict__ in, float* __restrict__ out, long total)
{   // (S,HW,C) -> (S,C,HW)
    const long idx = (long)blockIdx.x * 256 + threadIdx.x;
    if (idx >= total) return;
    const long s  = idx / ((long)HWSZ * CH);
    const long r  = idx - s * ((long)HWSZ * CH);
    const long c  = r / HWSZ;
    const long hw = r - c * HWSZ;
    out[idx] = in[(s * HWSZ + hw) * CH + c];
}

__global__ void cvt_f32_bf16(const float* __restrict__ in, __bf16* __restrict__ out, long n)
{
    const long idx = (long)blockIdx.x * 256 + threadIdx.x;
    if (idx < n) out[idx] = (__bf16)in[idx];
}

// ---------------------------------------------------------------------------
// Host orchestration
// ---------------------------------------------------------------------------
extern "C" void kernel_launch(void* const* d_in, const int* in_sizes, int n_in,
                              void* d_out, int out_size, void* d_ws, size_t ws_size,
                              hipStream_t stream)
{
    (void)in_sizes; (void)n_in; (void)out_size; (void)ws_size;
    const float* query = (const float*)d_in[0];
    const float* keyf  = (const float*)d_in[1];
    const float* qemb  = (const float*)d_in[2];
    const float* kemb  = (const float*)d_in[3];
    const float* Wv    = (const float*)d_in[4];
    const float* bv    = (const float*)d_in[5];
    const float* Wq    = (const float*)d_in[6];
    const float* bq    = (const float*)d_in[7];
    const float* Wk    = (const float*)d_in[8];
    const float* bk    = (const float*)d_in[9];
    const float* Wlp   = (const float*)d_in[10];
    const float* blp   = (const float*)d_in[11];
    const float* gam   = (const float*)d_in[12];
    const float* bet   = (const float*)d_in[13];
    const float* W1    = (const float*)d_in[14];
    const float* b1    = (const float*)d_in[15];
    const float* W2    = (const float*)d_in[16];
    const float* b2    = (const float*)d_in[17];

    const size_t TC = (size_t)TOKENS * CH;        // 16M elems
    float* xt  = (float*)d_ws;                    // current x, token-major (also MLP2 output)
    float* ktk = xt  + TC;                        // key token-major
    float* qb  = ktk + TC;                        // q proj / reused as yn
    float* kb  = qb  + TC;                        // k proj / reused as y
    float* vb  = kb  + TC;                        // v proj
    float* ob  = vb  + TC;                        // attn out (+lepe)
    float* hb  = ob  + TC;                        // MLP hidden, T x HIDDEN
    float* qet = hb  + (size_t)TOKENS * HIDDEN;   // D x HW x C
    float* ket = qet + (size_t)DEPTH * HWSZ * CH;
    __bf16* wqb = (__bf16*)(ket + (size_t)DEPTH * HWSZ * CH);
    __bf16* wkb = wqb + (size_t)DEPTH * CH * CH;
    __bf16* wvb = wkb + (size_t)DEPTH * CH * CH;
    __bf16* w1b = wvb + (size_t)DEPTH * CH * CH;
    __bf16* w2b = w1b + (size_t)DEPTH * HIDDEN * CH;

    // ---- layout prep ----
    {
        const long totB = (long)BATCH * CH * HWSZ;
        const long totD = (long)DEPTH * CH * HWSZ;
        chw_to_tc<<<(totB + 255) / 256, 256, 0, stream>>>(query, xt,  totB);
        chw_to_tc<<<(totB + 255) / 256, 256, 0, stream>>>(keyf,  ktk, totB);
        chw_to_tc<<<(totD + 255) / 256, 256, 0, stream>>>(qemb,  qet, totD);
        chw_to_tc<<<(totD + 255) / 256, 256, 0, stream>>>(kemb,  ket, totD);
        const long nw  = (long)DEPTH * CH * CH;
        const long nw1 = (long)DEPTH * HIDDEN * CH;
        cvt_f32_bf16<<<(nw  + 255) / 256, 256, 0, stream>>>(Wq, wqb, nw);
        cvt_f32_bf16<<<(nw  + 255) / 256, 256, 0, stream>>>(Wk, wkb, nw);
        cvt_f32_bf16<<<(nw  + 255) / 256, 256, 0, stream>>>(Wv, wvb, nw);
        cvt_f32_bf16<<<(nw1 + 255) / 256, 256, 0, stream>>>(W1, w1b, nw1);
        cvt_f32_bf16<<<(nw1 + 255) / 256, 256, 0, stream>>>(W2, w2b, nw1);
    }

    for (int d = 0; d < DEPTH; ++d) {
        const dim3 gProj(TOKENS / 128, CH / 256);      // (512,1)
        const dim3 gMlp1(TOKENS / 128, HIDDEN / 256);  // (512,4)
        // projections: q=(x+qe)Wq^T+bq, k=(key+ke)Wk^T+bk, v=key*Wv^T+bv
        gemm_bf16_wmma<true, false, false><<<gProj, 256, 0, stream>>>(
            xt,  qet + (size_t)d * HWSZ * CH,
            wqb + (size_t)d * CH * CH, bq + d * CH, nullptr, qb, TOKENS, CH, CH);
        gemm_bf16_wmma<true, false, false><<<gProj, 256, 0, stream>>>(
            ktk, ket + (size_t)d * HWSZ * CH,
            wkb + (size_t)d * CH * CH, bk + d * CH, nullptr, kb, TOKENS, CH, CH);
        gemm_bf16_wmma<false, false, false><<<gProj, 256, 0, stream>>>(
            ktk, nullptr,
            wvb + (size_t)d * CH * CH, bv + d * CH, nullptr, vb, TOKENS, CH, CH);
        // windowed attention + LePE
        attn_wmma<<<dim3(NWIN, HEADS), 64, 0, stream>>>(qb, kb, vb, ob);
        lepe_add<<<(unsigned)(TC / 256), 256, 0, stream>>>(vb, Wlp + (size_t)d * CH * 9,
                                                           blp + d * CH, ob);
        // residual + LN (y -> kb, yn -> qb; q/k projections are dead now)
        resid_layernorm<<<TOKENS / 8, 256, 0, stream>>>(xt, ob, gam + d * CH, bet + d * CH,
                                                        kb, qb);
        // MLP: h = gelu(yn*W1^T+b1); x' = y + h*W2^T + b2  (written back into xt)
        gemm_bf16_wmma<false, true, false><<<gMlp1, 256, 0, stream>>>(
            qb, nullptr,
            w1b + (size_t)d * HIDDEN * CH, b1 + d * HIDDEN, nullptr, hb, TOKENS, HIDDEN, CH);
        gemm_bf16_wmma<false, false, true><<<gProj, 256, 0, stream>>>(
            hb, nullptr,
            w2b + (size_t)d * CH * HIDDEN, b2 + d * CH, kb, xt, TOKENS, CH, HIDDEN);
    }

    // ---- back to (B,C,H,W) ----
    {
        const long totB = (long)BATCH * CH * HWSZ;
        tc_to_chw<<<(totB + 255) / 256, 256, 0, stream>>>(xt, (float*)d_out, totB);
    }
}
